// SwinTransformerBlock_60997125537802
// MI455X (gfx1250) — matrix operand order repaired and checked by reading, not compiled, not run on hardware
//
#include <hip/hip_runtime.h>
#include <hip/hip_bf16.h>

// ---------------------------------------------------------------------------
// Swin transformer block, MI455X (gfx1250, wave32, WMMA 16x16x32 f16).
//   * pad windows 49->64 tokens so every GEMM dim is WMMA-shaped
//   * f16 activations/weights, fp32 accumulate on v_wmma_f32_16x16x32_f16
//   * weights pre-swizzled to per-lane fragment layout -> B frags are two
//     global_load_b128 (no column gather, no v_mov_b16 packing)
//   * V stored transposed in LDS -> P@V B frags are two ds_load_b128, and the
//     producer writes packed 16B ds_store_b128
//   * LN1/shift/mask/bias/softmax/gelu/residuals fused around the tiles;
//     ~620MB total HBM traffic (~27us @ 23.3 TB/s) vs ~165 GFLOP compute.
// ---------------------------------------------------------------------------

typedef __attribute__((ext_vector_type(16))) _Float16 v16h;
typedef __attribute__((ext_vector_type(8)))  _Float16 v8h;
typedef __attribute__((ext_vector_type(8)))  float    v8f;

#define WMMA_F16(a, b, c) \
  __builtin_amdgcn_wmma_f32_16x16x32_f16(false, (a), false, (b), (short)0, (c), false, false)

// ---- fragment loaders (contiguous k = 16*(lane/16) + s in A and B, wave32) -------------
// A matrix 16x32 (MxK), row-major source, stride ld (halves).
// Per lane this is two contiguous 8-half runs -> 2x (ds|global)_load_b128.
__device__ __forceinline__ v16h frag_a(const _Float16* p, int ld) {
  const int lane = threadIdx.x & 31;
  const int row  = lane & 15;
  const int ko   = (lane >> 4) << 4;     // lanes 16..31 hold K+16..31 halves
  v16h a;
#pragma unroll
  for (int i = 0; i < 4; ++i) {
    const int k0 = 2 * i + ko;           // VGPR 0..3 : K = 0..7 (+ko)
    a[2 * i]     = p[row * ld + k0];
    a[2 * i + 1] = p[row * ld + k0 + 1];
    const int k1 = 2 * i + 8 + ko;       // VGPR 4..7 : K = 8..15 (+ko)
    a[2 * (i + 4)]     = p[row * ld + k1];
    a[2 * (i + 4) + 1] = p[row * ld + k1 + 1];
  }
  return a;
}

// B = M^T where M is NxK row-major (stride ld): b[k][n] = M[n][k].
// Contiguous 16 halves per lane -> 2x ds_load_b128.
__device__ __forceinline__ v16h frag_bT(const _Float16* p, int ld) {
  const int lane = threadIdx.x & 31;
  const int col  = lane & 15;
  const int k0   = (lane >> 4) << 4;
  v16h b;
#pragma unroll
  for (int j = 0; j < 16; ++j) b[j] = p[col * ld + k0 + j];
  return b;
}

// Swizzled B tile: 512 halves per 32x16 tile, lane-major (16 contiguous
// halves per lane). One aligned v16h load -> 2x global_load_b128.
__device__ __forceinline__ v16h frag_b_sw(const _Float16* tile) {
  return *(const v16h*)(tile + (threadIdx.x & 31) * 16);
}

__device__ __forceinline__ int regidx(int hs, int ws) {
  const int rh = hs < 49 ? 0 : (hs < 53 ? 1 : 2);
  const int rw = ws < 49 ? 0 : (ws < 53 ? 1 : 2);
  return rh * 3 + rw;
}

// ---------------------------------------------------------------------------
// Kernel 0: fp32 weight -> f16, swizzled into WMMA B-fragment layout.
// dst element e: j = e&15 (half within lane), l = (e>>4)&31 (lane),
// tile = e>>9, kt = tile % (K/32), nt = tile / (K/32).
// src element: row = kt*32 + ((l>>4)<<4) + j, col = nt*16 + (l&15).
// ---------------------------------------------------------------------------
__global__ void wswz_kernel(const float* __restrict__ src,
                            _Float16* __restrict__ dst, int Kdim, int Ndim) {
  const int e = blockIdx.x * 256 + threadIdx.x;
  if (e >= Kdim * Ndim) return;
  const int j    = e & 15;
  const int l    = (e >> 4) & 31;
  const int tile = e >> 9;
  const int KT   = Kdim >> 5;
  const int kt   = tile % KT;
  const int nt   = tile / KT;
  const int krow = kt * 32 + ((l >> 4) << 4) + j;
  const int ncol = nt * 16 + (l & 15);
  dst[e] = (_Float16)src[(size_t)krow * Ndim + ncol];
}

// ---------------------------------------------------------------------------
// Kernel 1: per-window fused  LN1 + shift + QKV + attention + proj + residual
// grid = 2048 windows, block = 256 threads (8 waves). Dynamic LDS ~163KB
// (exploits the 320KB WGP LDS; one workgroup per WGP).
// ---------------------------------------------------------------------------
#define LDA 264   // 256 + 8 pad (halves)
#define LDV 72    // V^T token stride  (halves)
#define LDP 68    // score row stride  (floats)
#define LDF 72    // prob  row stride  (halves)

__global__ __launch_bounds__(256)
void attn_block_kernel(const float* __restrict__ x,
                       const float* __restrict__ ln1g, const float* __restrict__ ln1b,
                       const _Float16* __restrict__ qkvw, const float* __restrict__ qkvb,
                       const float* __restrict__ rp,       // (169, 8)
                       const _Float16* __restrict__ projw, const float* __restrict__ projb,
                       float* __restrict__ y) {
  extern __shared__ char smem[];
  _Float16* Xs  = (_Float16*)smem;             // 64 x LDA (also reused as O)
  _Float16* Qs  = Xs + 64 * LDA;               // 64 x LDA
  _Float16* Ks  = Qs + 64 * LDA;               // 64 x LDA
  _Float16* VsT = Ks + 64 * LDA;               // 256 x LDV (V transposed: [d][tok])
  float*    Ps  = (float*)(VsT + 256 * LDV);   // 64 x LDP scores (one head)
  _Float16* Pf  = (_Float16*)(Ps + 64 * LDP);  // 64 x LDF probs f16
  float*  redS  = (float*)(Pf + 64 * LDF);     // 256
  float*  redQ  = redS + 256;                  // 256

  const int w    = blockIdx.x;
  const int b    = w >> 6;
  const int win  = w & 63;
  const int wr   = win >> 3, wc = win & 7;
  const int tid  = threadIdx.x;
  const int lane = tid & 31, wv = tid >> 5;
  const int ccol = lane & 15, rofs = (lane >> 4) * 8;

  // ---- LN1 + cyclic shift(-3,-3) + window gather -> f16 LDS, pad rows 0 ---
  {
    const int row = tid >> 2;         // 0..63 token
    const int prt = tid & 3;          // 64-wide column chunk
    const bool valid = row < 49;
    float v[64];
    float s = 0.f, q = 0.f;
    if (valid) {
      const int tr = row / 7, tc = row - (row / 7) * 7;
      int hsrc = wr * 7 + tr + 3; if (hsrc >= 56) hsrc -= 56;
      int wsrc = wc * 7 + tc + 3; if (wsrc >= 56) wsrc -= 56;
      const float* xr = x + (((size_t)b * 3136) + hsrc * 56 + wsrc) * 256 + prt * 64;
#pragma unroll
      for (int i = 0; i < 64; ++i) { const float t = xr[i]; v[i] = t; s += t; q += t * t; }
    }
    redS[tid] = s; redQ[tid] = q;
    __syncthreads();
    if (valid) {
      const float su = redS[row*4] + redS[row*4+1] + redS[row*4+2] + redS[row*4+3];
      const float sq = redQ[row*4] + redQ[row*4+1] + redQ[row*4+2] + redQ[row*4+3];
      const float mu = su * (1.f / 256.f);
      const float rs = rsqrtf(sq * (1.f / 256.f) - mu * mu + 1e-6f);
#pragma unroll
      for (int i = 0; i < 64; ++i) {
        const int cc = prt * 64 + i;
        Xs[row * LDA + cc] = (_Float16)((v[i] - mu) * rs * ln1g[cc] + ln1b[cc]);
      }
    } else {
#pragma unroll
      for (int i = 0; i < 64; ++i) Xs[row * LDA + prt * 64 + i] = (_Float16)0.f;
    }
  }
  __syncthreads();

  // ---- QKV GEMM: (64x256) @ (256x768) + bias, Q pre-scaled ----------------
  // qkvw swizzled: KT = 8, tile index = nt*8 + k.
  for (int t = wv; t < 192; t += 8) {
    const int mt = t & 3, nt = t >> 2;
    const int n0 = nt * 16;
    const float bv = qkvb[n0 + ccol];
    v8f acc;
#pragma unroll
    for (int r = 0; r < 8; ++r) acc[r] = bv;
#pragma unroll
    for (int k = 0; k < 8; ++k) {
      const v16h a  = frag_a(Xs + (mt * 16) * LDA + k * 32, LDA);
      const v16h bb = frag_b_sw(qkvw + (((size_t)nt * 8 + k) << 9));
      acc = WMMA_F16(a, bb, acc);
    }
    if (n0 < 512) {                        // Q, K: row-major token x d
      _Float16* dst = (n0 < 256) ? Qs : Ks;
      const int   cc = (n0 < 256) ? n0 : n0 - 256;
      const float sc = (n0 < 256) ? 0.17677669529663687f : 1.f;
#pragma unroll
      for (int r = 0; r < 8; ++r)
        dst[(mt * 16 + rofs + r) * LDA + cc + ccol] = (_Float16)(acc[r] * sc);
    } else {                               // V: transposed [d][tok], packed 16B
      const int cc = n0 - 512;
      v8h pk;
#pragma unroll
      for (int r = 0; r < 8; ++r) pk[r] = (_Float16)acc[r];
      *(v8h*)(VsT + (cc + ccol) * LDV + mt * 16 + rofs) = pk;
    }
  }
  __syncthreads();

  // ---- per-head attention: scores -> softmax -> P@V -----------------------
  for (int h = 0; h < 8; ++h) {
    // scores 64x64 = Qh(64x32) @ Kh^T, bias+mask in accumulator init
    for (int t = wv; t < 16; t += 8) {
      const int mt = t & 3, nt = t >> 2;
      const int kc = nt * 16 + ccol;               // key token
      v8f acc;
#pragma unroll
      for (int r = 0; r < 8; ++r) {
        const int qr = mt * 16 + rofs + r;         // query token
        float ini = -1e30f;                        // mask padded keys
        if (kc < 49) {
          if (qr < 49) {
            const int qh = qr / 7, qw = qr - qh * 7;
            const int kh = kc / 7, kw = kc - kh * 7;
            const float bias = rp[((qh - kh + 6) * 13 + (qw - kw + 6)) * 8 + h];
            const int rq = regidx(wr * 7 + qh, wc * 7 + qw);
            const int rk = regidx(wr * 7 + kh, wc * 7 + kw);
            ini = bias + ((rq != rk) ? -100.f : 0.f);
          } else {
            ini = 0.f;                             // padded query: harmless
          }
        }
        acc[r] = ini;
      }
      const v16h a  = frag_a (Qs + (mt * 16) * LDA + h * 32, LDA);
      const v16h bt = frag_bT(Ks + (nt * 16) * LDA + h * 32, LDA);
      acc = WMMA_F16(a, bt, acc);
#pragma unroll
      for (int r = 0; r < 8; ++r)
        Ps[(mt * 16 + rofs + r) * LDP + nt * 16 + ccol] = acc[r];
    }
    __syncthreads();

    // row softmax (64 rows)
    if (tid < 64) {
      float m = -1e30f;
#pragma unroll 8
      for (int j = 0; j < 64; ++j) m = fmaxf(m, Ps[tid * LDP + j]);
      float e[64]; float ss = 0.f;
#pragma unroll 8
      for (int j = 0; j < 64; ++j) { e[j] = __expf(Ps[tid * LDP + j] - m); ss += e[j]; }
      const float inv = 1.f / ss;
#pragma unroll 8
      for (int j = 0; j < 64; ++j) Pf[tid * LDF + j] = (_Float16)(e[j] * inv);
    }
    __syncthreads();

    // O_h (64x32) = P(64x64) @ V_h(64x32); 8 tiles, one per wave.
    // B[k][n] = V[tok k][d n] = VsT[n][k] -> contiguous frag_bT.
    {
      const int mt = wv & 3, nt = wv >> 2;
      v8f acc;
#pragma unroll
      for (int r = 0; r < 8; ++r) acc[r] = 0.f;
#pragma unroll
      for (int k = 0; k < 2; ++k) {
        const v16h a  = frag_a (Pf + (mt * 16) * LDF + k * 32, LDF);
        const v16h bb = frag_bT(VsT + (h * 32 + nt * 16) * LDV + k * 32, LDV);
        acc = WMMA_F16(a, bb, acc);
      }
#pragma unroll
      for (int r = 0; r < 8; ++r)       // O overlays Xs (Xs dead after QKV)
        Xs[(mt * 16 + rofs + r) * LDA + h * 32 + nt * 16 + ccol] = (_Float16)acc[r];
    }
    __syncthreads();
  }

  // ---- proj GEMM + un-shift + residual(x) -> y ----------------------------
  for (int t = wv; t < 64; t += 8) {
    const int mt = t & 3, nt = t >> 2;
    const int n0 = nt * 16;
    const float bv = projb[n0 + ccol];
    v8f acc;
#pragma unroll
    for (int r = 0; r < 8; ++r) acc[r] = bv;
#pragma unroll
    for (int k = 0; k < 8; ++k) {
      const v16h a  = frag_a(Xs + (mt * 16) * LDA + k * 32, LDA);
      const v16h bb = frag_b_sw(projw + (((size_t)nt * 8 + k) << 9));
      acc = WMMA_F16(a, bb, acc);
    }
#pragma unroll
    for (int r = 0; r < 8; ++r) {
      const int tok = mt * 16 + rofs + r;
      if (tok < 49) {
        const int tr = tok / 7, tc = tok - (tok / 7) * 7;
        int hf = wr * 7 + tr + 3; if (hf >= 56) hf -= 56;
        int wf = wc * 7 + tc + 3; if (wf >= 56) wf -= 56;
        const size_t idx = (((size_t)b * 3136) + hf * 56 + wf) * 256 + n0 + ccol;
        y[idx] = x[idx] + acc[r];          // y stays RT: re-read by MLP, fits L2
      }
    }
  }
}

// ---------------------------------------------------------------------------
// Kernel 2: fused  LN2 + MLP(gelu) + residual.  64-token tiles, 1568 blocks.
// Hidden (64x1024 f16 = 130KB) lives entirely in the 320KB WGP LDS.
// ---------------------------------------------------------------------------
#define LDH 1040  // 1024 + 16 pad (halves)

__global__ __launch_bounds__(256)
void mlp_block_kernel(const float* __restrict__ y,
                      const float* __restrict__ g2, const float* __restrict__ b2ln,
                      const _Float16* __restrict__ w1, const float* __restrict__ b1,
                      const _Float16* __restrict__ w2, const float* __restrict__ b2,
                      float* __restrict__ out) {
  extern __shared__ char smem[];
  _Float16* As = (_Float16*)smem;           // 64 x LDA (ln2 activations f16)
  _Float16* Hs = As + 64 * LDA;             // 64 x LDH gelu hidden f16
  float*  redS = (float*)(Hs + 64 * LDH);   // 256
  float*  redQ = redS + 256;                // 256

  const int tid  = threadIdx.x;
  const int lane = tid & 31, wv = tid >> 5;
  const int ccol = lane & 15, rofs = (lane >> 4) * 8;
  const size_t tok0 = (size_t)blockIdx.x * 64;

  // ---- LN2 on load -> f16 LDS --------------------------------------------
  {
    const int row = tid >> 2;
    const int prt = tid & 3;
    const float* yr = y + (tok0 + row) * 256 + prt * 64;
    float v[64]; float s = 0.f, q = 0.f;
#pragma unroll
    for (int i = 0; i < 64; ++i) { const float t = yr[i]; v[i] = t; s += t; q += t * t; }
    redS[tid] = s; redQ[tid] = q;
    __syncthreads();
    const float su = redS[row*4] + redS[row*4+1] + redS[row*4+2] + redS[row*4+3];
    const float sq = redQ[row*4] + redQ[row*4+1] + redQ[row*4+2] + redQ[row*4+3];
    const float mu = su * (1.f / 256.f);
    const float rs = rsqrtf(sq * (1.f / 256.f) - mu * mu + 1e-6f);
#pragma unroll
    for (int i = 0; i < 64; ++i) {
      const int cc = prt * 64 + i;
      As[row * LDA + cc] = (_Float16)((v[i] - mu) * rs * g2[cc] + b2ln[cc]);
    }
  }
  __syncthreads();

  // ---- stage 1: H = gelu(A @ w1 + b1), 64x1024 (w1 swizzled, KT=8) --------
  for (int t = wv; t < 256; t += 8) {
    const int mt = t & 3, nt = t >> 2;
    const int n0 = nt * 16;
    __builtin_prefetch(w1 + (((size_t)(((t + 8) >> 2) & 63) * 8) << 9), 0, 1);
    const float bv = b1[n0 + ccol];
    v8f acc;
#pragma unroll
    for (int r = 0; r < 8; ++r) acc[r] = bv;
#pragma unroll
    for (int k = 0; k < 8; ++k) {
      const v16h a  = frag_a(As + (mt * 16) * LDA + k * 32, LDA);
      const v16h bb = frag_b_sw(w1 + (((size_t)nt * 8 + k) << 9));
      acc = WMMA_F16(a, bb, acc);
    }
#pragma unroll
    for (int r = 0; r < 8; ++r) {
      const float xv = acc[r];
      const float g  = 0.5f * xv *
          (1.f + tanhf(0.7978845608028654f * (xv + 0.044715f * xv * xv * xv)));
      Hs[(mt * 16 + rofs + r) * LDH + n0 + ccol] = (_Float16)g;
    }
  }
  __syncthreads();

  // ---- stage 2: out = H @ w2 + b2 + y (w2 swizzled, KT=32) ----------------
  for (int t = wv; t < 64; t += 8) {
    const int mt = t & 3, nt = t >> 2;
    const int n0 = nt * 16;
    __builtin_prefetch(w2 + (((size_t)(((t + 8) >> 2) & 15) * 32) << 9), 0, 1);
    const float bv = b2[n0 + ccol];
    v8f acc;
#pragma unroll
    for (int r = 0; r < 8; ++r) acc[r] = bv;
#pragma unroll
    for (int k = 0; k < 32; ++k) {
      const v16h a  = frag_a(Hs + (mt * 16) * LDH + k * 32, LDH);
      const v16h bb = frag_b_sw(w2 + (((size_t)nt * 32 + k) << 9));
      acc = WMMA_F16(a, bb, acc);
    }
#pragma unroll
    for (int r = 0; r < 8; ++r) {
      const size_t idx = (tok0 + mt * 16 + rofs + r) * 256 + n0 + ccol;
      __builtin_nontemporal_store(y[idx] + acc[r], &out[idx]);  // NT: keep L2 for y/weights
    }
  }
}

// ---------------------------------------------------------------------------
extern "C" void kernel_launch(void* const* d_in, const int* in_sizes, int n_in,
                              void* d_out, int out_size, void* d_ws, size_t ws_size,
                              hipStream_t stream) {
  (void)in_sizes; (void)n_in; (void)out_size; (void)ws_size;
  const float* x     = (const float*)d_in[0];
  const float* ln1g  = (const float*)d_in[1];
  const float* ln1b  = (const float*)d_in[2];
  const float* qkvw  = (const float*)d_in[3];
  const float* qkvb  = (const float*)d_in[4];
  const float* rp    = (const float*)d_in[5];
  const float* projw = (const float*)d_in[6];
  const float* projb = (const float*)d_in[7];
  const float* ln2g  = (const float*)d_in[8];
  const float* ln2b  = (const float*)d_in[9];
  const float* w1    = (const float*)d_in[10];
  const float* b1    = (const float*)d_in[11];
  const float* w2    = (const float*)d_in[12];
  const float* b2    = (const float*)d_in[13];

  // workspace carve: swizzled f16 weights (~1.5MB, L2-resident) + y fp32 (103MB)
  char* ws = (char*)d_ws;
  _Float16* qkvw_s  = (_Float16*)ws; ws += (size_t)256 * 768  * 2;
  _Float16* projw_s = (_Float16*)ws; ws += (size_t)256 * 256  * 2;
  _Float16* w1_s    = (_Float16*)ws; ws += (size_t)256 * 1024 * 2;
  _Float16* w2_s    = (_Float16*)ws; ws += (size_t)1024 * 256 * 2;
  float* y = (float*)ws;                       // 32*3136*256 fp32

  wswz_kernel<<<(196608 + 255) / 256, 256, 0, stream>>>(qkvw, qkvw_s, 256, 768);
  wswz_kernel<<<(65536  + 255) / 256, 256, 0, stream>>>(projw, projw_s, 256, 256);
  wswz_kernel<<<(262144 + 255) / 256, 256, 0, stream>>>(w1, w1_s, 256, 1024);
  wswz_kernel<<<(262144 + 255) / 256, 256, 0, stream>>>(w2, w2_s, 1024, 256);

  const size_t sm_attn = (size_t)3 * 64 * LDA * sizeof(_Float16)   // Xs,Qs,Ks
                       + (size_t)256 * LDV * sizeof(_Float16)      // VsT
                       + (size_t)64 * LDP * sizeof(float)          // Ps
                       + (size_t)64 * LDF * sizeof(_Float16)       // Pf
                       + 2 * 256 * sizeof(float);                  // reductions
  attn_block_kernel<<<2048, 256, sm_attn, stream>>>(
      x, ln1g, ln1b, qkvw_s, qkvb, rp, projw_s, projb, y);

  const size_t sm_mlp = (size_t)64 * LDA * sizeof(_Float16)        // As
                      + (size_t)64 * LDH * sizeof(_Float16)        // Hs
                      + 2 * 256 * sizeof(float);
  mlp_block_kernel<<<1568, 256, sm_mlp, stream>>>(
      y, ln2g, ln2b, w1_s, b1, w2_s, b2, (float*)d_out);
}
